// WassersteinBarycenter_42691974922398
// MI455X (gfx1250) — matrix operand order repaired and checked by reading, not compiled, not run on hardware
//
#include <hip/hip_runtime.h>
#include <hip/hip_bf16.h>

typedef __attribute__((ext_vector_type(16))) _Float16 v16h;
typedef __attribute__((ext_vector_type(8)))  _Float16 v8h;
typedef __attribute__((ext_vector_type(4)))  _Float16 v4h;
typedef __attribute__((ext_vector_type(8)))  float    v8f;
typedef __attribute__((ext_vector_type(8)))  unsigned short v8u;

#define NBINS       10
#define NGROUPS     4
#define HIST_BLOCKS 1024
#define HIST_THRDS  256
#define WAVES_PB    (HIST_THRDS / 32)
#define TILE        128                 // samples per wave per iteration

// ---------------------------------------------------------------------------
// Zero the workspace accumulators (harness poisons d_ws once, never again).
// ---------------------------------------------------------------------------
__global__ void ws_zero_kernel(float* __restrict__ ws) {
    if (threadIdx.x < 64) ws[threadIdx.x] = 0.0f;
}

// Fast sigmoid: v_exp_f32 + v_rcp_f32 (loss scalar does not need IEEE division)
__device__ __forceinline__ float fast_sigmoid(float x) {
    const float t = __builtin_amdgcn_exp2f(-1.44269504f * x);
    return __builtin_amdgcn_rcpf(1.0f + t);
}

// Packed-f16 |v| : clear both sign bits per dword (4x v_and_b32 per v16h)
__device__ __forceinline__ v16h vabs16(v16h v) {
    v8u u;
    __builtin_memcpy(&u, &v, sizeof(u));
    #pragma unroll
    for (int i = 0; i < 8; ++i) u[i] &= 0x7FFFu;
    v16h r;
    __builtin_memcpy(&r, &u, sizeof(r));
    return r;
}

// ---------------------------------------------------------------------------
// Soft-histogram as a WMMA GEMM:
//   counts[g, bin] = sum_k onehot(label_k == g) * relu(0.1 - |c_k - 0.1*bin|)
// Wave processes 128 samples/iter (float4/int4 loads), stages c and labels as
// f16 in LDS, builds both operands with packed-f16 pipelines, and issues 4
// chained v_wmma_f32_16x16x32_f16. One-hot is computed arithmetically:
// max(1 - |label - row|, 0), exact for small integers in f16 (sentinel -1 -> 0).
// ---------------------------------------------------------------------------
__global__ __launch_bounds__(HIST_THRDS)
void ws_hist_wmma_kernel(const float* __restrict__ acts,
                         const int*   __restrict__ labels,
                         float*       __restrict__ counts,
                         int N) {
    __shared__ _Float16 s_c[WAVES_PB][TILE];
    __shared__ _Float16 s_l[WAVES_PB][TILE];

    const int lane = threadIdx.x & 31;
    const int wv   = threadIdx.x >> 5;
    const int half = lane >> 4;     // which 16-lane half of the wave
    const int n    = lane & 15;     // B column (bin) / A row (group id)

    // Broadcast constant vectors (packed-f16 operands)
    v16h locv, nv, rv, onev, zerov;
    #pragma unroll
    for (int t = 0; t < 16; ++t) {
        locv[t]  = (_Float16)(0.1f * (float)n);
        nv[t]    = (_Float16)(float)n;
        rv[t]    = (_Float16)0.1f;
        onev[t]  = (_Float16)1.0f;
        zerov[t] = (_Float16)0.0f;
    }

    const int wavesTotal = gridDim.x * WAVES_PB;
    const int waveId     = blockIdx.x * WAVES_PB + wv;
    const int stride     = wavesTotal * TILE;

    v8f acc = {0.f, 0.f, 0.f, 0.f, 0.f, 0.f, 0.f, 0.f};

    for (int base = waveId * TILE; base < N; base += stride) {
        float4 x4;
        int4   g4;
        const bool fullTile = (base + TILE <= N);   // wave-uniform branch
        if (fullTile) {
            x4 = ((const float4*)(acts   + base))[lane];
            g4 = ((const int4*  )(labels + base))[lane];
        } else {
            const int s0 = base + lane * 4;
            x4.x = acts[min(s0 + 0, N - 1)];
            x4.y = acts[min(s0 + 1, N - 1)];
            x4.z = acts[min(s0 + 2, N - 1)];
            x4.w = acts[min(s0 + 3, N - 1)];
            g4.x = (s0 + 0 < N) ? labels[s0 + 0] : -1;
            g4.y = (s0 + 1 < N) ? labels[s0 + 1] : -1;
            g4.z = (s0 + 2 < N) ? labels[s0 + 2] : -1;
            g4.w = (s0 + 3 < N) ? labels[s0 + 3] : -1;
        }
        // Speculative next-sweep prefetch (dropped if OOB).
        __builtin_prefetch(acts   + base + stride + lane * 4, 0, 1);
        __builtin_prefetch(labels + base + stride + lane * 4, 0, 1);

        float c0 = fast_sigmoid(x4.x) - 0.0001f;
        float c1 = fast_sigmoid(x4.y) - 0.0001f;
        float c2 = fast_sigmoid(x4.z) - 0.0001f;
        float c3 = fast_sigmoid(x4.w) - 0.0001f;
        if (!fullTile) {                 // sentinel: f16 inf -> zero weight
            const int s0 = base + lane * 4;
            if (s0 + 0 >= N) c0 = 1.0e9f;
            if (s0 + 1 >= N) c1 = 1.0e9f;
            if (s0 + 2 >= N) c2 = 1.0e9f;
            if (s0 + 3 >= N) c3 = 1.0e9f;
        }

        // Stage the wave's 128 samples in LDS as f16 (two ds_store_b64).
        v4h cpk = {(_Float16)c0, (_Float16)c1, (_Float16)c2, (_Float16)c3};
        v4h lpk = {(_Float16)g4.x, (_Float16)g4.y,
                   (_Float16)g4.z, (_Float16)g4.w};
        *(v4h*)&s_c[wv][lane * 4] = cpk;
        *(v4h*)&s_l[wv][lane * 4] = lpk;
        asm volatile("s_wait_dscnt 0" ::: "memory");  // wave-synchronous LDS RAW

        #pragma unroll
        for (int tile = 0; tile < 4; ++tile) {
            // B operand: lane holds column n; element t <-> K-row = half*16+t
            // (contiguous 32B -> 2x ds_load_b128), then 4 pk-f16 ops per pair.
            v16h cv = *(const v16h*)&s_c[wv][tile * 32 + half * 16];
            v16h d  = vabs16(cv - locv);
            v16h bm = __builtin_elementwise_max(rv - d, zerov);

            // A operand: lane holds row m = n; same K order as B:
            //   t = 0..15 -> K = half*16 + t (contiguous 32B)
            v8h llo = *(const v8h*)&s_l[wv][tile * 32 + half * 16];
            v8h lhi = *(const v8h*)&s_l[wv][tile * 32 + half * 16 + 8];
            v16h lv = __builtin_shufflevector(llo, lhi,
                        0, 1, 2, 3, 4, 5, 6, 7,
                        8, 9, 10, 11, 12, 13, 14, 15);
            v16h e  = vabs16(lv - nv);
            v16h am = __builtin_elementwise_max(onev - e, zerov);

            acc = __builtin_amdgcn_wmma_f32_16x16x32_f16(
                /*neg_a=*/false, am, /*neg_b=*/false, bm,
                /*c_mod=*/(short)0, acc, /*reuse_a=*/false, /*reuse_b=*/false);
        }
    }

    // C/D layout: lanes 0-15, VGPR v <-> (M=v, N=lane). Groups = rows 0..3,
    // bins = columns 0..9 -> lanes 0..9 flush VGPRs 0..3.
    if (lane < NBINS) {
        #pragma unroll
        for (int g = 0; g < NGROUPS; ++g)
            atomicAdd(&counts[g * NBINS + lane], acc[g]);
    }
}

// ---------------------------------------------------------------------------
// Finalize: normalize histograms, 1-D W2 vs bary_est on support 0..9,
// replicating jnp.searchsorted(side='left') + clip semantics. Trivial size.
// ---------------------------------------------------------------------------
__global__ void ws_finalize_kernel(const float* __restrict__ counts,
                                   const float* __restrict__ bary,
                                   float* __restrict__ out) {
    const int tid = threadIdx.x;
    if (tid < NBINS) out[1 + tid] = bary[tid];   // bary_est pass-through

    if (tid == 0) {
        float cb[NBINS];
        float accb = 0.0f;
        for (int j = 0; j < NBINS; ++j) { accb += bary[j]; cb[j] = accb; }

        float loss = 0.0f;
        for (int g = 0; g < NGROUPS; ++g) {
            float h[NBINS];
            float sum = 0.0f;
            for (int j = 0; j < NBINS; ++j) {
                h[j] = counts[g * NBINS + j] + 0.0001f;
                sum += h[j];
            }
            const float inv = 1.0f / sum;   // double-normalize is a no-op
            float ca[NBINS];
            float acca = 0.0f;
            for (int j = 0; j < NBINS; ++j) { acca += h[j] * inv; ca[j] = acca; }

            // Merge the two sorted CDFs -> 20 quantile levels.
            float qs[2 * NBINS];
            int ia = 0, ib = 0;
            for (int i = 0; i < 2 * NBINS; ++i) {
                const float va = (ia < NBINS) ? ca[ia] : 3.0e38f;
                const float vb = (ib < NBINS) ? cb[ib] : 3.0e38f;
                if (va <= vb) { qs[i] = va; ++ia; } else { qs[i] = vb; ++ib; }
            }

            float prev = 0.0f;
            for (int i = 0; i < 2 * NBINS; ++i) {
                const float q = qs[i];
                int xa = 0; while (xa < NBINS && ca[xa] < q) ++xa;  // side='left'
                int xb = 0; while (xb < NBINS && cb[xb] < q) ++xb;
                if (xa > NBINS - 1) xa = NBINS - 1;                 // clip
                if (xb > NBINS - 1) xb = NBINS - 1;
                const float d = (float)(xa - xb);
                loss += (q - prev) * d * d;                         // p = 2
                prev = q;
            }
        }
        out[0] = loss;
    }
}

// ---------------------------------------------------------------------------
// d_in: [0]=acts f32[N], [1]=bary_est f32[10], [2]=group_labels i32[N],
//       [3]=nbins (10), [4]=n_groups (4)   (constants baked in)
// d_out: 11 floats = [loss, bary_est...]
// ---------------------------------------------------------------------------
extern "C" void kernel_launch(void* const* d_in, const int* in_sizes, int n_in,
                              void* d_out, int out_size, void* d_ws, size_t ws_size,
                              hipStream_t stream) {
    const float* acts   = (const float*)d_in[0];
    const float* bary   = (const float*)d_in[1];
    const int*   labels = (const int*)d_in[2];
    const int    N      = in_sizes[0];

    float* counts = (float*)d_ws;   // 64 floats used
    float* out    = (float*)d_out;

    ws_zero_kernel<<<1, 64, 0, stream>>>(counts);
    ws_hist_wmma_kernel<<<HIST_BLOCKS, HIST_THRDS, 0, stream>>>(acts, labels, counts, N);
    ws_finalize_kernel<<<1, 32, 0, stream>>>(counts, bary, out);
}
